// InteractiveGCN_44641890075195
// MI455X (gfx1250) — matrix operand order repaired and checked by reading, not compiled, not run on hardware
//
#include <hip/hip_runtime.h>
#include <cstdint>

// ---------------- problem constants (match reference) ----------------
#define N_ENT 30000
#define NE    200000
#define NEA   400000
#define NR    500
#define EHD   300
#define RHD   100
#define NH    8
#define H2    2
#define OUTC  1000
#define X2C   500
#define EPSF  1e-16f

// ---------------- WMMA types ----------------
typedef __attribute__((ext_vector_type(16))) _Float16 v16h;
typedef __attribute__((ext_vector_type(8)))  _Float16 v8h;
typedef __attribute__((ext_vector_type(4)))  _Float16 v4h;
typedef __attribute__((ext_vector_type(8)))  float    v8f;

union V16H { v16h v; v8h p[2]; _Float16 h[16]; };
union V8F  { v8f  v; float    f[8];  };
union V4H  { v4h  v; _Float16 h[4];  };

// ordered-uint encoding for float atomic max
__device__ __forceinline__ unsigned f2ord(float f) {
  unsigned b = __float_as_uint(f);
  return (b & 0x80000000u) ? ~b : (b | 0x80000000u);
}
__device__ __forceinline__ float ord2f(unsigned u) {
  return (u & 0x80000000u) ? __uint_as_float(u & 0x7fffffffu) : __uint_as_float(~u);
}
__device__ __forceinline__ float lrelu(float v) { return v > 0.f ? v : 0.01f * v; }

// =====================================================================
// WMMA GEMM: out[M,Nout] = A[M,K] * W[Nout,K]^T  (+bias, sigmoid if act==1)
// One wave -> 16(M) x 32(N) of D: one A fragment feeds two
// v_wmma_f32_16x16x32_f16 per 32-wide K step. Tiles staged in per-wave LDS
// (f32 -> f16 on the fly); interior tiles use unpredicated float4 loads and
// 8-byte packed LDS stores; operand fragments are contiguous -> ds_load_b128.
// =====================================================================
#define WPB 4
__global__ __launch_bounds__(128)
void k_wmma_gemm(const float* __restrict__ A, int lda,
                 const float* __restrict__ W,
                 const float* __restrict__ bias,
                 float* __restrict__ out, int ldo,
                 int M, int K, int Nout, int act)
{
  // row pitch 40 halves (80 B) keeps every 8/16-half chunk 16-B aligned
  __shared__ __align__(16) _Float16 As[WPB][16][40];
  __shared__ __align__(16) _Float16 Bs[WPB][32][40];

  const int wave = threadIdx.x >> 5;
  const int lane = threadIdx.x & 31;
  const int tm = blockIdx.x * WPB + wave;
  const int m0 = tm * 16;
  const int n0 = blockIdx.y * 32;
  if (m0 >= M) return;                    // wave-uniform exit; no barriers used

  const int g  = lane >> 4;               // half-wave group
  const int ml = lane & 15;

  V8F acc0, acc1;
#pragma unroll
  for (int i = 0; i < 8; ++i) { acc0.f[i] = 0.f; acc1.f[i] = 0.f; }

  for (int k0 = 0; k0 < K; k0 += 32) {
    // ---- stage A tile: As[r][c] = A[m0+r][k0+c], 16x32 ----
    if (m0 + 16 <= M && k0 + 32 <= K) {
#pragma unroll
      for (int it = 0; it < 4; ++it) {    // 128 float4 quads / 32 lanes
        int idx = it * 32 + lane;
        int r = idx >> 3, c4 = (idx & 7) * 4;
        const float* src = A + (size_t)(m0 + r) * lda + k0 + c4;
        float4 vv = *reinterpret_cast<const float4*>(src);
        V4H pk;
        pk.h[0] = (_Float16)vv.x; pk.h[1] = (_Float16)vv.y;
        pk.h[2] = (_Float16)vv.z; pk.h[3] = (_Float16)vv.w;
        *reinterpret_cast<v4h*>(&As[wave][r][c4]) = pk.v;
        if (k0 + 32 < K) __builtin_prefetch(src + 32, 0, 0);
      }
    } else {
      for (int idx = lane; idx < 512; idx += 32) {
        int r = idx >> 5, c = idx & 31;
        int mm = m0 + r, kk = k0 + c;
        As[wave][r][c] = (mm < M && kk < K) ? (_Float16)A[(size_t)mm * lda + kk]
                                            : (_Float16)0.f;
      }
    }
    // ---- stage B tile: Bs[n][k] = W[n0+n][k0+k], 32x32 ----
    if (n0 + 32 <= Nout && k0 + 32 <= K) {
#pragma unroll
      for (int it = 0; it < 8; ++it) {    // 256 quads / 32 lanes
        int idx = it * 32 + lane;
        int r = idx >> 3, c4 = (idx & 7) * 4;
        const float* src = W + (size_t)(n0 + r) * K + k0 + c4;
        float4 vv = *reinterpret_cast<const float4*>(src);
        V4H pk;
        pk.h[0] = (_Float16)vv.x; pk.h[1] = (_Float16)vv.y;
        pk.h[2] = (_Float16)vv.z; pk.h[3] = (_Float16)vv.w;
        *reinterpret_cast<v4h*>(&Bs[wave][r][c4]) = pk.v;
      }
    } else {
      for (int idx = lane; idx < 1024; idx += 32) {
        int r = idx >> 5, c = idx & 31;
        int nn = n0 + r, kk = k0 + c;
        Bs[wave][r][c] = (nn < Nout && kk < K) ? (_Float16)W[(size_t)nn * K + kk]
                                               : (_Float16)0.f;
      }
    }
    // ---- fragments: contiguous LDS rows -> ds_load_b128 pairs ----
    V16H a, b0, b1;
    a.p[0]  = *reinterpret_cast<const v8h*>(&As[wave][ml][16 * g]);      // K 16g..16g+7
    a.p[1]  = *reinterpret_cast<const v8h*>(&As[wave][ml][16 * g + 8]);  // K 16g+8..
    b0.p[0] = *reinterpret_cast<const v8h*>(&Bs[wave][ml][16 * g]);      // N tile 0
    b0.p[1] = *reinterpret_cast<const v8h*>(&Bs[wave][ml][16 * g + 8]);
    b1.p[0] = *reinterpret_cast<const v8h*>(&Bs[wave][16 + ml][16 * g]); // N tile 1
    b1.p[1] = *reinterpret_cast<const v8h*>(&Bs[wave][16 + ml][16 * g + 8]);

    acc0.v = __builtin_amdgcn_wmma_f32_16x16x32_f16(
        false, a.v, false, b0.v, (short)0, acc0.v, false, false);
    acc1.v = __builtin_amdgcn_wmma_f32_16x16x32_f16(
        false, a.v, false, b1.v, (short)0, acc1.v, false, false);
  }

  // D layout: VGPR r -> M = m0 + r + 8g, N = n0 + ml (+16 for tile 1)
#pragma unroll
  for (int half = 0; half < 2; ++half) {
    int n = n0 + 16 * half + ml;
    if (n >= Nout) continue;
    const V8F& acc = half ? acc1 : acc0;
#pragma unroll
    for (int r = 0; r < 8; ++r) {
      int m = m0 + r + 8 * g;
      if (m < M) {
        float v = acc.f[r];
        if (act == 1) v = 1.f / (1.f + expf(-(v + bias[n])));
        out[(size_t)m * ldo + n] = v;
      }
    }
  }
}

// =====================================================================
// elementwise / fill
// =====================================================================
__global__ void k_fill_f32(float* p, float v, int n) {
  int t = blockIdx.x * blockDim.x + threadIdx.x;
  if (t < n) p[t] = v;
}
__global__ void k_fill_u32(unsigned* p, unsigned v, int n) {
  int t = blockIdx.x * blockDim.x + threadIdx.x;
  if (t < n) p[t] = v;
}

// degree count over target nodes i = ei[1]
__global__ void k_deg(const int* __restrict__ ei, float* deg, int E) {
  int t = blockIdx.x * blockDim.x + threadIdx.x;
  if (t < E) atomicAdd(&deg[ei[E + t]], 1.0f);
}
__global__ void k_invsqrt(float* d, int n) {
  int t = blockIdx.x * blockDim.x + threadIdx.x;
  if (t < n) { float v = d[t]; d[t] = v > 0.f ? rsqrtf(v) : 0.f; }
}

// GCN aggregation: thread per (edge, 4-channel chunk); EH=300 -> 75 chunks
__global__ void k_gcn_agg(const float* __restrict__ x, const int* __restrict__ ei,
                          const float* __restrict__ dis, float* __restrict__ agg, int E) {
  const int CH = EHD / 4;
  int t = blockIdx.x * blockDim.x + threadIdx.x;
  int e = t / CH, q = t - e * CH;
  if (e >= E) return;
  int j = ei[e], i = ei[E + e];
  float nrm = dis[j] * dis[i];
  if (nrm == 0.f) return;
  const float4 v = *reinterpret_cast<const float4*>(x + (size_t)j * EHD + q * 4);
  float* dst = agg + (size_t)i * EHD + q * 4;
  atomicAdd(dst + 0, nrm * v.x);
  atomicAdd(dst + 1, nrm * v.y);
  atomicAdd(dst + 2, nrm * v.z);
  atomicAdd(dst + 3, nrm * v.w);
}

// highway: out = gate*relu(agg) + (1-gate)*x1
__global__ void k_highway(const float* __restrict__ x1, const float* __restrict__ agg,
                          const float* __restrict__ gate, float* __restrict__ out, int n) {
  int t = blockIdx.x * blockDim.x + threadIdx.x;
  if (t >= n) return;
  float g = gate[t];
  float a = agg[t]; a = a > 0.f ? a : 0.f;
  out[t] = g * a + (1.f - g) * x1[t];
}

// =====================================================================
// generic segment-softmax attention (3 passes + per-edge weight)
// =====================================================================
__global__ void k_att_logit_max(const float* __restrict__ sS, const float* __restrict__ sD,
                                const int* __restrict__ gS, const int* __restrict__ gD,
                                const int* __restrict__ seg,
                                float* __restrict__ la, unsigned* __restrict__ m,
                                int E, int Hc) {
  int t = blockIdx.x * blockDim.x + threadIdx.x;
  if (t >= E * Hc) return;
  int e = t / Hc, h = t - e * Hc;
  float v = sS[(size_t)gS[e] * Hc + h] + sD[(size_t)gD[e] * Hc + h];
  v = lrelu(v);
  la[t] = v;
  atomicMax(&m[(size_t)seg[e] * Hc + h], f2ord(v));
}
__global__ void k_att_expsum(float* __restrict__ la, const unsigned* __restrict__ m,
                             float* __restrict__ s, const int* __restrict__ seg,
                             int E, int Hc) {
  int t = blockIdx.x * blockDim.x + threadIdx.x;
  if (t >= E * Hc) return;
  int e = t / Hc, h = t - e * Hc;
  float mx = ord2f(m[(size_t)seg[e] * Hc + h]);
  float ex = expf(la[t] - mx);
  la[t] = ex;
  atomicAdd(&s[(size_t)seg[e] * Hc + h], ex);
}
__global__ void k_att_weight(const float* __restrict__ la, const float* __restrict__ s,
                             const int* __restrict__ seg, float* __restrict__ w,
                             int E, int Hc, float scale) {
  int e = blockIdx.x * blockDim.x + threadIdx.x;
  if (e >= E) return;
  int sg = seg[e];
  float sum = 0.f;
  for (int h = 0; h < Hc; ++h)
    sum += la[(size_t)e * Hc + h] / (s[(size_t)sg * Hc + h] + EPSF);
  w[e] = sum * scale;
}
__global__ void k_att_scatter(const float* __restrict__ w, const float* __restrict__ feat,
                              int ldf, const int* __restrict__ gf, const int* __restrict__ seg,
                              float* __restrict__ out, int ldo, int E, int F) {
  int CH = F >> 2;
  int t = blockIdx.x * blockDim.x + threadIdx.x;
  int e = t / CH, q = t - e * CH;
  if (e >= E) return;
  float ww = w[e];
  const float4 v = *reinterpret_cast<const float4*>(feat + (size_t)gf[e] * ldf + q * 4);
  float* d = out + (size_t)seg[e] * ldo + q * 4;
  atomicAdd(d + 0, ww * v.x);
  atomicAdd(d + 1, ww * v.y);
  atomicAdd(d + 2, ww * v.z);
  atomicAdd(d + 3, ww * v.w);
}

// x_r = 0.25*(xr_h + xr_t)   ( /H_E2R /2 )
__global__ void k_combine_xr(const float* a, const float* b, float* o, int n) {
  int t = blockIdx.x * blockDim.x + threadIdx.x;
  if (t < n) o[t] = 0.25f * (a[t] + b[t]);
}

// write x2 = concat(x, xe_h, xe_t) into out[:, :500]; zero out[:, 500:]
__global__ void k_write_x2(const float* __restrict__ x, const float* __restrict__ xeh,
                           const float* __restrict__ xet, float* __restrict__ out, int n) {
  int t = blockIdx.x * blockDim.x + threadIdx.x;
  if (t >= n * OUTC) return;
  int row = t / OUTC, c = t - row * OUTC;
  float v;
  if (c < EHD)            v = x[(size_t)row * EHD + c];
  else if (c < EHD + RHD) v = xeh[(size_t)row * RHD + (c - EHD)];
  else if (c < X2C)       v = xet[(size_t)row * RHD + (c - EHD - RHD)];
  else                    v = 0.f;
  out[t] = v;
}

// =====================================================================
// host launcher
// =====================================================================
extern "C" void kernel_launch(void* const* d_in, const int* in_sizes, int n_in,
                              void* d_out, int out_size, void* d_ws, size_t ws_size,
                              hipStream_t stream) {
  (void)in_sizes; (void)n_in; (void)out_size; (void)ws_size;
  const float* x_e    = (const float*)d_in[0];
  const int*   ei     = (const int*)d_in[1];    // [2,E]   row0=h, row1=t
  const int*   rel    = (const int*)d_in[2];
  const int*   eall   = (const int*)d_in[3];    // [2,EA]  row0=j, row1=i
  const float* hw1_w  = (const float*)d_in[5];
  const float* hw1_b  = (const float*)d_in[6];
  const float* hw2_w  = (const float*)d_in[7];
  const float* hw2_b  = (const float*)d_in[8];
  const float* e2r_wh = (const float*)d_in[9];
  const float* e2r_wt = (const float*)d_in[10];
  const float* e2r_ah = (const float*)d_in[11];
  const float* e2r_at = (const float*)d_in[12];
  const float* r2e_ah = (const float*)d_in[13];
  const float* r2e_at = (const float*)d_in[14];
  const float* r2e_ar = (const float*)d_in[15];
  const float* gat_ai = (const float*)d_in[16];
  const float* gat_aj = (const float*)d_in[17];
  float* out = (float*)d_out;

  const int* h_idx = ei;           const int* t_idx = ei + NE;
  const int* j_a   = eall;         const int* i_a   = eall + NEA;

  float* ws = (float*)d_ws;
  // region layout (float offsets): dis | A(9M) | G(9M) | x0(9M) | x1(9M)  ~144MB
  float* dis = ws + 0;
  float* RA  = ws + 32768;
  float* RG  = RA + 9000000;
  float* x0  = RG + 9000000;
  float* x1  = x0 + 9000000;
  float* agg = RA;                 // phase-1 alias of region A
  float* gat = RG;                 // phase-1 alias of region G (gate buffer)

  // region A, phase 2 (E2R)
  float*    xh  = RA;              float*    xt  = RA + 3000000;
  float*    sh1 = RA + 6000000;    float*    st1 = RA + 6060000;
  float*    sh2 = RA + 6120000;    float*    st2 = RA + 6180000;
  float*    la1 = RA + 6240000;    float*    la2 = RA + 6640000;
  unsigned* m1  = (unsigned*)(RA + 7040000);  float* s1 = RA + 7042048;
  unsigned* m2  = (unsigned*)(RA + 7044096);  float* s2 = RA + 7046144;
  float*    xrh = RA + 7050000;    float*    xrt = RA + 7100000;
  float*    x_r = RA + 7150000;
  float*    w1  = RA + 7200000;    float*    w2  = RA + 7400000;
  // region A, phase 3 reuse (xh/xt dead)
  float*    xeh = RA;              float*    xet = RA + 3000000;
  // region G, phase 3 (R2E)
  float*    seh = RG;              float*    set = RG + 240000;
  float*    sr  = RG + 480000;
  float*    lah = RG + 500000;     float*    lat = RG + 2100000;
  unsigned* mh  = (unsigned*)(RG + 3700000);  float* sh = RG + 3940000;
  unsigned* mt  = (unsigned*)(RG + 4180000);  float* st = RG + 4420000;
  float*    wh  = RG + 4660000;    float*    wt = RG + 4860000;
  // region A, phase 4 (GAT) reuse
  float*    si  = RA;              float*    sj = RA + 240000;
  unsigned* mg  = (unsigned*)(RA + 480000);   float* sg = RA + 720000;
  float*    lag = RA + 1000000;    float*    wg = RA + 4200000;

  const unsigned ORD_NEG_INF = 0x007FFFFFu;  // f2ord(-inf)
  auto cdiv = [](int a, int b) { return (a + b - 1) / b; };
  const int TB = 256;
  auto fillf = [&](float* p, float v, int n) {
    k_fill_f32<<<cdiv(n, TB), TB, 0, stream>>>(p, v, n);
  };
  auto fillu = [&](unsigned* p, unsigned v, int n) {
    k_fill_u32<<<cdiv(n, TB), TB, 0, stream>>>(p, v, n);
  };
  auto gemm = [&](const float* A, int lda, const float* W, const float* bias,
                  float* o, int ldo, int M, int K, int Nout, int act) {
    dim3 g(cdiv(cdiv(M, 16), WPB), cdiv(Nout, 32));
    k_wmma_gemm<<<g, dim3(128), 0, stream>>>(A, lda, W, bias, o, ldo, M, K, Nout, act);
  };

  // ---------- degree normalization (shared by both GCNs) ----------
  fillf(dis, 0.f, N_ENT);
  k_deg<<<cdiv(NEA, TB), TB, 0, stream>>>(eall, dis, NEA);
  k_invsqrt<<<cdiv(N_ENT, TB), TB, 0, stream>>>(dis, N_ENT);

  // ---------- GCN1 + Highway1 ----------
  fillf(agg, 0.f, N_ENT * EHD);
  k_gcn_agg<<<cdiv(NEA * (EHD / 4), TB), TB, 0, stream>>>(x_e, eall, dis, agg, NEA);
  gemm(x_e, EHD, hw1_w, hw1_b, gat, EHD, N_ENT, EHD, EHD, 1);     // gate=sigmoid(x@W^T+b)
  k_highway<<<cdiv(N_ENT * EHD, TB), TB, 0, stream>>>(x_e, agg, gat, x0, N_ENT * EHD);

  // ---------- GCN2 + Highway2 ----------
  fillf(agg, 0.f, N_ENT * EHD);
  k_gcn_agg<<<cdiv(NEA * (EHD / 4), TB), TB, 0, stream>>>(x0, eall, dis, agg, NEA);
  gemm(x0, EHD, hw2_w, hw2_b, gat, EHD, N_ENT, EHD, EHD, 1);
  k_highway<<<cdiv(N_ENT * EHD, TB), TB, 0, stream>>>(x0, agg, gat, x1, N_ENT * EHD);

  // ---------- E2R projections ----------
  gemm(x1, EHD, e2r_wh, nullptr, xh, RHD, N_ENT, EHD, RHD, 0);    // xh = x@wh^T
  gemm(x1, EHD, e2r_wt, nullptr, xt, RHD, N_ENT, EHD, RHD, 0);
  gemm(xh, RHD, e2r_ah, nullptr, sh1, H2, N_ENT, RHD, H2, 0);     // (xh@ah^T)
  gemm(xt, RHD, e2r_at, nullptr, st1, H2, N_ENT, RHD, H2, 0);     // (xt@at^T)
  gemm(xt, RHD, e2r_ah, nullptr, sh2, H2, N_ENT, RHD, H2, 0);     // (xt@ah^T)
  gemm(xh, RHD, e2r_at, nullptr, st2, H2, N_ENT, RHD, H2, 0);     // (xh@at^T)

  // ---------- E2R segment softmax over rel (2 heads, both directions) ----------
  fillu(m1, ORD_NEG_INF, NR * H2);  fillf(s1, 0.f, NR * H2);
  fillu(m2, ORD_NEG_INF, NR * H2);  fillf(s2, 0.f, NR * H2);
  fillf(xrh, 0.f, NR * RHD);        fillf(xrt, 0.f, NR * RHD);
  k_att_logit_max<<<cdiv(NE * H2, TB), TB, 0, stream>>>(sh1, st1, h_idx, t_idx, rel, la1, m1, NE, H2);
  k_att_expsum  <<<cdiv(NE * H2, TB), TB, 0, stream>>>(la1, m1, s1, rel, NE, H2);
  k_att_weight  <<<cdiv(NE, TB), TB, 0, stream>>>(la1, s1, rel, w1, NE, H2, 1.0f);
  k_att_scatter <<<cdiv(NE * (RHD / 4), TB), TB, 0, stream>>>(w1, xh, RHD, h_idx, rel, xrh, RHD, NE, RHD);
  k_att_logit_max<<<cdiv(NE * H2, TB), TB, 0, stream>>>(sh2, st2, h_idx, t_idx, rel, la2, m2, NE, H2);
  k_att_expsum  <<<cdiv(NE * H2, TB), TB, 0, stream>>>(la2, m2, s2, rel, NE, H2);
  k_att_weight  <<<cdiv(NE, TB), TB, 0, stream>>>(la2, s2, rel, w2, NE, H2, 1.0f);
  k_att_scatter <<<cdiv(NE * (RHD / 4), TB), TB, 0, stream>>>(w2, xt, RHD, t_idx, rel, xrt, RHD, NE, RHD);
  k_combine_xr  <<<cdiv(NR * RHD, TB), TB, 0, stream>>>(xrh, xrt, x_r, NR * RHD);

  // ---------- R2E (8 heads) ----------
  gemm(x1, EHD, r2e_ah, nullptr, seh, NH, N_ENT, EHD, NH, 0);
  gemm(x1, EHD, r2e_at, nullptr, set, NH, N_ENT, EHD, NH, 0);
  gemm(x_r, RHD, r2e_ar, nullptr, sr, NH, NR, RHD, NH, 0);
  fillu(mh, ORD_NEG_INF, N_ENT * NH);  fillf(sh, 0.f, N_ENT * NH);
  fillu(mt, ORD_NEG_INF, N_ENT * NH);  fillf(st, 0.f, N_ENT * NH);
  fillf(xeh, 0.f, N_ENT * RHD);        fillf(xet, 0.f, N_ENT * RHD);
  k_att_logit_max<<<cdiv(NE * NH, TB), TB, 0, stream>>>(seh, sr, h_idx, rel, h_idx, lah, mh, NE, NH);
  k_att_expsum  <<<cdiv(NE * NH, TB), TB, 0, stream>>>(lah, mh, sh, h_idx, NE, NH);
  k_att_weight  <<<cdiv(NE, TB), TB, 0, stream>>>(lah, sh, h_idx, wh, NE, NH, 1.0f / NH);
  k_att_scatter <<<cdiv(NE * (RHD / 4), TB), TB, 0, stream>>>(wh, x_r, RHD, rel, h_idx, xeh, RHD, NE, RHD);
  k_att_logit_max<<<cdiv(NE * NH, TB), TB, 0, stream>>>(set, sr, t_idx, rel, t_idx, lat, mt, NE, NH);
  k_att_expsum  <<<cdiv(NE * NH, TB), TB, 0, stream>>>(lat, mt, st, t_idx, NE, NH);
  k_att_weight  <<<cdiv(NE, TB), TB, 0, stream>>>(lat, st, t_idx, wt, NE, NH, 1.0f / NH);
  k_att_scatter <<<cdiv(NE * (RHD / 4), TB), TB, 0, stream>>>(wt, x_r, RHD, rel, t_idx, xet, RHD, NE, RHD);

  // ---------- x2 -> out[:, :500]; zero out[:, 500:] ----------
  k_write_x2<<<cdiv(N_ENT * OUTC, TB), TB, 0, stream>>>(x1, xeh, xet, out, N_ENT);

  // ---------- GAT over full graph (8 heads), reading x2 from out ----------
  gemm(out, OUTC, gat_ai, nullptr, si, NH, N_ENT, X2C, NH, 0);    // (x2@ai^T)
  gemm(out, OUTC, gat_aj, nullptr, sj, NH, N_ENT, X2C, NH, 0);
  fillu(mg, ORD_NEG_INF, N_ENT * NH);  fillf(sg, 0.f, N_ENT * NH);
  k_att_logit_max<<<cdiv(NEA * NH, TB), TB, 0, stream>>>(si, sj, i_a, j_a, i_a, lag, mg, NEA, NH);
  k_att_expsum  <<<cdiv(NEA * NH, TB), TB, 0, stream>>>(lag, mg, sg, i_a, NEA, NH);
  k_att_weight  <<<cdiv(NEA, TB), TB, 0, stream>>>(lag, sg, i_a, wg, NEA, NH, 1.0f / NH);
  k_att_scatter <<<cdiv(NEA * (X2C / 4), TB), TB, 0, stream>>>(wg, out, OUTC, j_a, i_a,
                                                              out + X2C, OUTC, NEA, X2C);
}
